// LoraFluxAttnProcessor_50440095924232
// MI455X (gfx1250) — matrix operand order repaired and checked by reading, not compiled, not run on hardware
//
#include <hip/hip_runtime.h>

// ---------------------------------------------------------------------------
// Types
// ---------------------------------------------------------------------------
typedef _Float16 v16h __attribute__((ext_vector_type(16)));
typedef _Float16 v8h  __attribute__((ext_vector_type(8)));
typedef _Float16 v4h  __attribute__((ext_vector_type(4)));
typedef float    v8f  __attribute__((ext_vector_type(8)));

#define L_SEQ 2048
#define DIM   3072
#define NH    24
#define HD    128
#define QKV_N (3 * DIM)          // 9216
#define RANK_PAD 32              // rank 16 padded to 32 for one K=32 WMMA
#define LORA_SCALE 1.0f          // alpha/rank*weight = 16/16*1

// ---------------------------------------------------------------------------
// WMMA helpers (f16 inputs, f32 accumulate) — CDNA5 16x16x32
// ---------------------------------------------------------------------------
static __device__ __forceinline__ v8f wmma16(v16h a, v16h b, v8f c) {
  return __builtin_amdgcn_wmma_f32_16x16x32_f16(false, a, false, b,
                                                (short)0, c, false, false);
}

// A-matrix 16x32 fragment. `row` points at (row m, K-chunk base).
// Regs 0-3 hold K=[khalf*8, +8), regs 4-7 hold K=[16+khalf*8, +8).
static __device__ __forceinline__ v16h load_afrag(const _Float16* row,
                                                  unsigned khalf) {
  v8h lo = *(const v8h*)(row + khalf * 8);
  v8h hi = *(const v8h*)(row + 16 + khalf * 8);
  return __builtin_shufflevector(lo, hi, 0, 1, 2, 3, 4, 5, 6, 7,
                                 8, 9, 10, 11, 12, 13, 14, 15);
}

// B-matrix 32x16 fragment. `row` points at (col n's K-row, chunk base).
// Regs 0-3 hold K=[kg*8, +8), regs 4-7 hold K=[16+kg*8, +8).
static __device__ __forceinline__ v16h load_bfrag(const _Float16* row,
                                                  unsigned kg) {
  v8h lo = *(const v8h*)(row + kg * 8);
  v8h hi = *(const v8h*)(row + 16 + kg * 8);
  return __builtin_shufflevector(lo, hi, 0, 1, 2, 3, 4, 5, 6, 7,
                                 8, 9, 10, 11, 12, 13, 14, 15);
}

// Async global -> LDS, 16 bytes per lane. IOFFSET applies to both sides.
static __device__ __forceinline__ void async_g2l_b128(unsigned lds_off,
                                                      const void* gptr) {
  asm volatile("global_load_async_to_lds_b128 %0, %1, off"
               :: "v"(lds_off),
                  "v"((unsigned long long)(__UINTPTR_TYPE__)gptr)
               : "memory");
}
static __device__ __forceinline__ void async_g2l_b128_o16(unsigned lds_off,
                                                          const void* gptr) {
  asm volatile("global_load_async_to_lds_b128 %0, %1, off offset:16"
               :: "v"(lds_off),
                  "v"((unsigned long long)(__UINTPTR_TYPE__)gptr)
               : "memory");
}
static __device__ __forceinline__ void wait_async4() {
  asm volatile("s_wait_asynccnt 0x4" ::: "memory");
}
static __device__ __forceinline__ void wait_async0() {
  asm volatile("s_wait_asynccnt 0x0" ::: "memory");
}

// Reductions across the 16-lane half-groups (C-matrix rows live in one half).
static __device__ __forceinline__ float rowmax16(float v) {
  v = fmaxf(v, __shfl_xor(v, 1, 32));
  v = fmaxf(v, __shfl_xor(v, 2, 32));
  v = fmaxf(v, __shfl_xor(v, 4, 32));
  v = fmaxf(v, __shfl_xor(v, 8, 32));
  return v;
}
static __device__ __forceinline__ float rowsum16(float v) {
  v += __shfl_xor(v, 1, 32);
  v += __shfl_xor(v, 2, 32);
  v += __shfl_xor(v, 4, 32);
  v += __shfl_xor(v, 8, 32);
  return v;
}

// ---------------------------------------------------------------------------
// Elementwise converts
// ---------------------------------------------------------------------------
__global__ void f32_to_f16_kernel(const float* __restrict__ s,
                                  _Float16* __restrict__ d, long n) {
  long i = (long)blockIdx.x * blockDim.x + threadIdx.x;
  long stride = (long)gridDim.x * blockDim.x;
  for (; i < n; i += stride) d[i] = (_Float16)s[i];
}

// up [N,16] fp32 -> [N,32] f16, cols 16..31 zero
__global__ void pad_up_kernel(const float* __restrict__ s,
                              _Float16* __restrict__ d, int N) {
  long n = (long)N * RANK_PAD;
  long i = (long)blockIdx.x * blockDim.x + threadIdx.x;
  long stride = (long)gridDim.x * blockDim.x;
  for (; i < n; i += stride) {
    int row = (int)(i >> 5), c = (int)(i & 31);
    d[i] = (c < 16) ? (_Float16)s[(long)row * 16 + c] : (_Float16)0.0f;
  }
}

// ---------------------------------------------------------------------------
// LoRA down: T[M,32] = scale * (A[M,K] @ Dn[16,K]^T), cols 16..31 zeroed
// One wave per 16-row tile, 8 waves/block.
// ---------------------------------------------------------------------------
__global__ __launch_bounds__(256)
void lora_down_kernel(const _Float16* __restrict__ A,
                      const _Float16* __restrict__ Dn,
                      _Float16* __restrict__ T, int M, int K, float scale) {
  const int lane = threadIdx.x & 31, wid = threadIdx.x >> 5;
  const unsigned nlane = lane & 15, khalf = (unsigned)(lane >> 4);
  const int m0 = (blockIdx.x * 8 + wid) * 16;
  if (m0 >= M) return;
  v8f acc = {};
  const _Float16* arow = A + (size_t)(m0 + nlane) * K;
  const _Float16* brow = Dn + (size_t)nlane * K;
  for (int k0 = 0; k0 < K; k0 += 32) {
    v16h af = load_afrag(arow + k0, khalf);
    v16h bf = load_bfrag(brow + k0, khalf);
    acc = wmma16(af, bf, acc);
  }
#pragma unroll
  for (int r = 0; r < 8; ++r) {
    int row = m0 + (int)khalf * 8 + r;
    T[(size_t)row * RANK_PAD + nlane] = (_Float16)(acc[r] * scale);
    T[(size_t)row * RANK_PAD + nlane + 16] = (_Float16)0.0f;
  }
}

// ---------------------------------------------------------------------------
// Tiled WMMA GEMM: C[M,N] = A[M,K] @ Bw[N,K]^T + bias + tA[M,32] @ tB[N,32]^T
// 128x128 block tile, K chunks of 32. Double-buffered LDS filled by
// global_load_async_to_lds_b128 (ASYNCcnt), overlapping DMA with WMMAs.
// Per stage each thread issues 4 async b128 loads (2 for A, 2 for B), so
// s_wait_asynccnt 4 retires exactly the previous stage.
// ---------------------------------------------------------------------------
__global__ __launch_bounds__(256)
void gemm_f16_kernel(const _Float16* __restrict__ A,
                     const _Float16* __restrict__ Bw,
                     const float* __restrict__ bias,
                     const _Float16* __restrict__ tA,
                     const _Float16* __restrict__ tB,
                     float* __restrict__ C, int M, int N, int K) {
  __shared__ __align__(16) _Float16 As[2][128 * 48];
  __shared__ __align__(16) _Float16 Bs[2][128 * 48];
  const int tid = threadIdx.x, lane = tid & 31, wid = tid >> 5;
  const unsigned nlane = lane & 15, khalf = (unsigned)(lane >> 4);
  const int bm = blockIdx.y * 128, bn = blockIdx.x * 128;
  const int wm = (wid >> 1) * 32, wn = (wid & 1) * 64;
  const int lrow = tid >> 1, lseg = tid & 1;

  const _Float16* gA = A + (size_t)(bm + lrow) * K + lseg * 16;
  const _Float16* gB = Bw + (size_t)(bn + lrow) * K + lseg * 16;
  const unsigned ldsA0 =
      (unsigned)(__UINTPTR_TYPE__)&As[0][lrow * 48 + lseg * 16];
  const unsigned ldsA1 =
      (unsigned)(__UINTPTR_TYPE__)&As[1][lrow * 48 + lseg * 16];
  const unsigned ldsB0 =
      (unsigned)(__UINTPTR_TYPE__)&Bs[0][lrow * 48 + lseg * 16];
  const unsigned ldsB1 =
      (unsigned)(__UINTPTR_TYPE__)&Bs[1][lrow * 48 + lseg * 16];

  v8f acc[2][4] = {};

  const int niters = K / 32;
  // prologue: fill buffer 0
  async_g2l_b128(ldsA0, gA);
  async_g2l_b128_o16(ldsA0, gA);
  async_g2l_b128(ldsB0, gB);
  async_g2l_b128_o16(ldsB0, gB);

  for (int it = 0; it < niters; ++it) {
    const int buf = it & 1;
    if (it + 1 < niters) {
      const _Float16* nA = gA + (it + 1) * 32;
      const _Float16* nB = gB + (it + 1) * 32;
      const unsigned la = buf ? ldsA0 : ldsA1;
      const unsigned lb = buf ? ldsB0 : ldsB1;
      async_g2l_b128(la, nA);
      async_g2l_b128_o16(la, nA);
      async_g2l_b128(lb, nB);
      async_g2l_b128_o16(lb, nB);
      wait_async4();  // current stage (older 4 per thread) retired
    } else {
      wait_async0();
    }
    __syncthreads();

    v16h bf[4];
#pragma unroll
    for (int nt = 0; nt < 4; ++nt)
      bf[nt] = load_bfrag(&Bs[buf][(wn + nt * 16 + nlane) * 48], khalf);
#pragma unroll
    for (int mt = 0; mt < 2; ++mt) {
      v16h af = load_afrag(&As[buf][(wm + mt * 16 + nlane) * 48], khalf);
#pragma unroll
      for (int nt = 0; nt < 4; ++nt)
        acc[mt][nt] = wmma16(af, bf[nt], acc[mt][nt]);
    }
    __syncthreads();  // safe to refill this buffer two stages later
  }

  // LoRA epilogue: one extra K=32 WMMA per tile (upper 16 K are zero-padded)
  if (tA != nullptr) {
#pragma unroll
    for (int mt = 0; mt < 2; ++mt) {
      v16h af = load_afrag(tA + (size_t)(bm + wm + mt * 16 + nlane) * RANK_PAD,
                           khalf);
#pragma unroll
      for (int nt = 0; nt < 4; ++nt) {
        v16h bf = load_bfrag(tB + (size_t)(bn + wn + nt * 16 + nlane) * RANK_PAD,
                             khalf);
        acc[mt][nt] = wmma16(af, bf, acc[mt][nt]);
      }
    }
  }

#pragma unroll
  for (int mt = 0; mt < 2; ++mt) {
#pragma unroll
    for (int nt = 0; nt < 4; ++nt) {
      int col = bn + wn + nt * 16 + (int)nlane;
      float bv = bias ? bias[col] : 0.0f;
      int r0 = bm + wm + mt * 16 + (int)khalf * 8;
#pragma unroll
      for (int r = 0; r < 8; ++r)
        C[(size_t)(r0 + r) * N + col] = acc[mt][nt][r] + bv;
    }
  }
}

// ---------------------------------------------------------------------------
// Per-head RMSNorm + RoPE + scatter to attention layouts.
// qkv fp32 [L, 9216]; pe fp32 [L, 64, 2, 2];
// outputs: q,k f16 [H, L, 128]; vT f16 [H, 128, L]
// One wave per (h, l) row; each lane owns 4 contiguous d.
// ---------------------------------------------------------------------------
__global__ __launch_bounds__(256)
void qknorm_rope_kernel(const float* __restrict__ qkv,
                        const float* __restrict__ pe,
                        const float* __restrict__ qs,
                        const float* __restrict__ ks,
                        _Float16* __restrict__ qh, _Float16* __restrict__ kh,
                        _Float16* __restrict__ vT) {
  const int lane = threadIdx.x & 31, wid = threadIdx.x >> 5;
  const int idx = blockIdx.x * 8 + wid;  // 0 .. H*L-1
  const int h = idx >> 11, l = idx & (L_SEQ - 1);
  const int d0 = lane * 4;
  const float* base = qkv + (size_t)l * QKV_N + h * HD + d0;

  float4 qv = *(const float4*)(base);
  float4 kv = *(const float4*)(base + DIM);
  float4 vv = *(const float4*)(base + 2 * DIM);

  float sq = qv.x * qv.x + qv.y * qv.y + qv.z * qv.z + qv.w * qv.w;
  float sk = kv.x * kv.x + kv.y * kv.y + kv.z * kv.z + kv.w * kv.w;
#pragma unroll
  for (int m = 16; m >= 1; m >>= 1) {
    sq += __shfl_xor(sq, m, 32);
    sk += __shfl_xor(sk, m, 32);
  }
  float rq = rsqrtf(sq * (1.0f / HD) + 1e-6f);
  float rk = rsqrtf(sk * (1.0f / HD) + 1e-6f);

  float4 qsc = *(const float4*)(qs + d0);
  float4 ksc = *(const float4*)(ks + d0);

  float q0 = qv.x * rq * qsc.x, q1 = qv.y * rq * qsc.y;
  float q2 = qv.z * rq * qsc.z, q3 = qv.w * rq * qsc.w;
  float k0 = kv.x * rk * ksc.x, k1 = kv.y * rk * ksc.y;
  float k2 = kv.z * rk * ksc.z, k3 = kv.w * rk * ksc.w;

  // pe[l, d0/2 .. d0/2+1, 2, 2] = 8 contiguous floats
  const float* pp = pe + (size_t)l * 256 + (size_t)d0 * 2;
  float4 p0 = *(const float4*)(pp);
  float4 p1 = *(const float4*)(pp + 4);

  v4h qo, ko;
  qo.x = (_Float16)(p0.x * q0 + p0.y * q1);
  qo.y = (_Float16)(p0.z * q0 + p0.w * q1);
  qo.z = (_Float16)(p1.x * q2 + p1.y * q3);
  qo.w = (_Float16)(p1.z * q2 + p1.w * q3);
  ko.x = (_Float16)(p0.x * k0 + p0.y * k1);
  ko.y = (_Float16)(p0.z * k0 + p0.w * k1);
  ko.z = (_Float16)(p1.x * k2 + p1.y * k3);
  ko.w = (_Float16)(p1.z * k2 + p1.w * k3);

  size_t ro = ((size_t)h * L_SEQ + l) * HD + d0;
  *(v4h*)(qh + ro) = qo;
  *(v4h*)(kh + ro) = ko;

  vT[((size_t)h * HD + d0 + 0) * L_SEQ + l] = (_Float16)vv.x;
  vT[((size_t)h * HD + d0 + 1) * L_SEQ + l] = (_Float16)vv.y;
  vT[((size_t)h * HD + d0 + 2) * L_SEQ + l] = (_Float16)vv.z;
  vT[((size_t)h * HD + d0 + 3) * L_SEQ + l] = (_Float16)vv.w;
}

// ---------------------------------------------------------------------------
// Flash attention per head. 8 waves/block, each wave owns a 16-row Q tile;
// streams 64-key blocks with online softmax; P staged through LDS to go from
// C-matrix layout to A-fragment layout for the PV WMMAs.
// ---------------------------------------------------------------------------
__global__ __launch_bounds__(256)
void attn_kernel(const _Float16* __restrict__ qh,
                 const _Float16* __restrict__ kh,
                 const _Float16* __restrict__ vT,
                 _Float16* __restrict__ oh) {
  __shared__ __align__(16) _Float16 Pb[8][16 * 80];  // per-wave 16x64, stride 80
  const int lane = threadIdx.x & 31, wid = threadIdx.x >> 5;
  const unsigned nlane = lane & 15, khalf = (unsigned)(lane >> 4);
  const int h = blockIdx.x >> 4;       // 16 blocks per head
  const int m0 = (blockIdx.x & 15) * 128 + wid * 16;

  // Q fragments for d = 0..127 (4 K-chunks), kept resident
  const _Float16* qrow = qh + ((size_t)h * L_SEQ + m0 + nlane) * HD;
  v16h qf[4];
#pragma unroll
  for (int kc = 0; kc < 4; ++kc) qf[kc] = load_afrag(qrow + kc * 32, khalf);

  v8f Oacc[8] = {};
  float mo[8], li[8];
#pragma unroll
  for (int r = 0; r < 8; ++r) { mo[r] = -1e30f; li[r] = 0.0f; }

  const float sc = 0.0883883476483184f;  // 1/sqrt(128)

  for (int j = 0; j < L_SEQ; j += 64) {
    v8f S[4] = {};
#pragma unroll
    for (int nt = 0; nt < 4; ++nt) {
      const _Float16* krow =
          kh + ((size_t)h * L_SEQ + j + nt * 16 + nlane) * HD;
#pragma unroll
      for (int kc = 0; kc < 4; ++kc) {
        v16h bf = load_bfrag(krow + kc * 32, khalf);
        S[nt] = wmma16(qf[kc], bf, S[nt]);
      }
    }

    // online softmax
#pragma unroll
    for (int nt = 0; nt < 4; ++nt)
#pragma unroll
      for (int r = 0; r < 8; ++r) S[nt][r] *= sc;

    float al[8];
#pragma unroll
    for (int r = 0; r < 8; ++r) {
      float v = fmaxf(fmaxf(S[0][r], S[1][r]), fmaxf(S[2][r], S[3][r]));
      float mn = fmaxf(mo[r], rowmax16(v));
      al[r] = __expf(mo[r] - mn);
      mo[r] = mn;
    }
#pragma unroll
    for (int nt = 0; nt < 4; ++nt)
#pragma unroll
      for (int r = 0; r < 8; ++r) S[nt][r] = __expf(S[nt][r] - mo[r]);
#pragma unroll
    for (int r = 0; r < 8; ++r) {
      float rs = rowsum16(S[0][r] + S[1][r] + S[2][r] + S[3][r]);
      li[r] = li[r] * al[r] + rs;
    }
#pragma unroll
    for (int vt = 0; vt < 8; ++vt)
#pragma unroll
      for (int r = 0; r < 8; ++r) Oacc[vt][r] *= al[r];

    // P (C layout) -> LDS (row major 16x64, stride 80)
#pragma unroll
    for (int nt = 0; nt < 4; ++nt)
#pragma unroll
      for (int r = 0; r < 8; ++r)
        Pb[wid][((int)khalf * 8 + r) * 80 + nt * 16 + (int)nlane] =
            (_Float16)S[nt][r];
    // same-wave LDS ops are in-order: store->load hazard handled by DScnt waits

    v16h pf[2];
    const _Float16* prow = &Pb[wid][(int)nlane * 80];
#pragma unroll
    for (int kc = 0; kc < 2; ++kc) pf[kc] = load_afrag(prow + kc * 32, khalf);

#pragma unroll
    for (int vt = 0; vt < 8; ++vt) {
      const _Float16* vrow =
          vT + ((size_t)h * HD + vt * 16 + nlane) * L_SEQ + j;
#pragma unroll
      for (int kc = 0; kc < 2; ++kc) {
        v16h bf = load_bfrag(vrow + kc * 32, khalf);
        Oacc[vt] = wmma16(pf[kc], bf, Oacc[vt]);
      }
    }
  }

  // finalize and write o in [L, H*HD] f16 for the proj GEMM
#pragma unroll
  for (int vt = 0; vt < 8; ++vt) {
    int col = h * HD + vt * 16 + (int)nlane;
#pragma unroll
    for (int r = 0; r < 8; ++r) {
      int row = m0 + (int)khalf * 8 + r;
      oh[(size_t)row * DIM + col] = (_Float16)(Oacc[vt][r] / li[r]);
    }
  }
}

// ---------------------------------------------------------------------------
// Host launcher
// ---------------------------------------------------------------------------
extern "C" void kernel_launch(void* const* d_in, const int* in_sizes, int n_in,
                              void* d_out, int out_size, void* d_ws,
                              size_t ws_size, hipStream_t stream) {
  (void)in_sizes; (void)n_in; (void)out_size; (void)ws_size;
  const float* x        = (const float*)d_in[0];
  const float* pe       = (const float*)d_in[1];
  const float* qkv_w    = (const float*)d_in[2];
  const float* qkv_b    = (const float*)d_in[3];
  const float* proj_w   = (const float*)d_in[4];
  const float* proj_b   = (const float*)d_in[5];
  const float* qkv_dn   = (const float*)d_in[6];
  const float* qkv_up   = (const float*)d_in[7];
  const float* proj_dn  = (const float*)d_in[8];
  const float* proj_up  = (const float*)d_in[9];
  const float* q_scale  = (const float*)d_in[10];
  const float* k_scale  = (const float*)d_in[11];
  float* out = (float*)d_out;

  // carve workspace
  char* p = (char*)d_ws;
  auto carve = [&](size_t bytes) {
    void* r = (void*)p;
    p += (bytes + 255) & ~(size_t)255;
    return r;
  };
  _Float16* xh   = (_Float16*)carve((size_t)L_SEQ * DIM * 2);
  _Float16* wqh  = (_Float16*)carve((size_t)QKV_N * DIM * 2);
  _Float16* wph  = (_Float16*)carve((size_t)DIM * DIM * 2);
  _Float16* dnq  = (_Float16*)carve((size_t)16 * DIM * 2);
  _Float16* dnp  = (_Float16*)carve((size_t)16 * DIM * 2);
  _Float16* upq  = (_Float16*)carve((size_t)QKV_N * RANK_PAD * 2);
  _Float16* upp  = (_Float16*)carve((size_t)DIM * RANK_PAD * 2);
  _Float16* t1   = (_Float16*)carve((size_t)L_SEQ * RANK_PAD * 2);
  _Float16* t2   = (_Float16*)carve((size_t)L_SEQ * RANK_PAD * 2);
  float*    qkv32 = (float*)carve((size_t)L_SEQ * QKV_N * 4);
  _Float16* qhb  = (_Float16*)carve((size_t)NH * L_SEQ * HD * 2);
  _Float16* khb  = (_Float16*)carve((size_t)NH * L_SEQ * HD * 2);
  _Float16* vTb  = (_Float16*)carve((size_t)NH * HD * L_SEQ * 2);
  _Float16* ohb  = (_Float16*)carve((size_t)L_SEQ * DIM * 2);

  auto conv = [&](const float* s, _Float16* d, long n) {
    int g = (int)((n + 4095) / 4096);
    if (g > 2048) g = 2048;
    f32_to_f16_kernel<<<g, 256, 0, stream>>>(s, d, n);
  };
  conv(x, xh, (long)L_SEQ * DIM);
  conv(qkv_w, wqh, (long)QKV_N * DIM);
  conv(proj_w, wph, (long)DIM * DIM);
  conv(qkv_dn, dnq, (long)16 * DIM);
  conv(proj_dn, dnp, (long)16 * DIM);
  pad_up_kernel<<<256, 256, 0, stream>>>(qkv_up, upq, QKV_N);
  pad_up_kernel<<<128, 256, 0, stream>>>(proj_up, upp, DIM);

  // LoRA down for QKV: t1 = scale * x @ down^T
  lora_down_kernel<<<L_SEQ / 128, 256, 0, stream>>>(xh, dnq, t1, L_SEQ, DIM,
                                                    LORA_SCALE);
  // QKV GEMM + bias + LoRA epilogue
  {
    dim3 grid(QKV_N / 128, L_SEQ / 128);
    gemm_f16_kernel<<<grid, 256, 0, stream>>>(xh, wqh, qkv_b, t1, upq, qkv32,
                                              L_SEQ, QKV_N, DIM);
  }
  // RMSNorm + RoPE + scatter
  qknorm_rope_kernel<<<NH * L_SEQ / 8, 256, 0, stream>>>(
      qkv32, pe, q_scale, k_scale, qhb, khb, vTb);
  // Attention
  attn_kernel<<<NH * 16, 256, 0, stream>>>(qhb, khb, vTb, ohb);
  // LoRA down for proj: t2 = scale * o @ down^T
  lora_down_kernel<<<L_SEQ / 128, 256, 0, stream>>>(ohb, dnp, t2, L_SEQ, DIM,
                                                    LORA_SCALE);
  // Output proj GEMM + bias + LoRA epilogue -> d_out (fp32)
  {
    dim3 grid(DIM / 128, L_SEQ / 128);
    gemm_f16_kernel<<<grid, 256, 0, stream>>>(ohb, wph, proj_b, t2, upp, out,
                                              L_SEQ, DIM, DIM);
  }
}
